// MixtureOfExpertsLayer_24670292148715
// MI455X (gfx1250) — compile-verified, matrix operand order checked by reading, not run on hardware
//
#include <hip/hip_runtime.h>
#include <math.h>

typedef __bf16 bf16_t;
typedef __attribute__((ext_vector_type(16))) __bf16 v16bf;
typedef __attribute__((ext_vector_type(8)))  float  v8f;
typedef __attribute__((ext_vector_type(4)))  unsigned int v4u;
typedef __attribute__((ext_vector_type(8)))  int v8i;
typedef __attribute__((ext_vector_type(4)))  int v4i;

#define M_TOT 8192
#define K_DIM 1024
#define N_DIM 1024
#define E_NUM 8
#define BM 128
#define BN 128
#define BK 64
#define ASTR 72                          // 64 elems + 8 pad (144 B rows, 16B aligned)
#define BSTR 72                          // TDM pad: 32 DWORDs data + 4 DWORDs pad
#define KB_PER_E (K_DIM / BK)            // 16
#define NKB (KB_PER_E * E_NUM)           // 128 fused-K stages (2 WMMA k-steps each)

#if defined(__AMDGCN__) && __has_builtin(__builtin_amdgcn_tensor_load_to_lds) && \
    __has_builtin(__builtin_amdgcn_s_wait_tensorcnt)
#define USE_TDM 1
#else
#define USE_TDM 0
#endif

// ------------- expert_w f32 [E][K][N] -> bf16 transposed [E][N][K] -------------
__global__ __launch_bounds__(256) void moe_convert_w_t(const float* __restrict__ w,
                                                       bf16_t* __restrict__ o) {
    __shared__ bf16_t t[32][34];
    const int e  = blockIdx.z;
    const int k0 = blockIdx.y * 32;
    const int n0 = blockIdx.x * 32;
    const int c  = threadIdx.x & 31;
    const int r0 = threadIdx.x >> 5;     // 0..7
    const float* src = w + ((size_t)e * K_DIM + k0) * N_DIM + n0;
#pragma unroll
    for (int rr = 0; rr < 32; rr += 8)
        t[r0 + rr][c] = (bf16_t)src[(size_t)(r0 + rr) * N_DIM + c];
    __syncthreads();
    bf16_t* dst = o + ((size_t)e * N_DIM + n0) * K_DIM + k0;
#pragma unroll
    for (int rr = 0; rr < 32; rr += 8)
        dst[(size_t)(r0 + rr) * K_DIM + c] = t[c][r0 + rr];
}

// ---------------- gating: one wave32 per row, softmax over E ----------------
__global__ __launch_bounds__(256) void moe_gate(const float* __restrict__ x,
                                                const float* __restrict__ gw,
                                                const float* __restrict__ gb,
                                                float* __restrict__ gate) {
    const int row  = blockIdx.x * 8 + (threadIdx.x >> 5);
    const int lane = threadIdx.x & 31;
    float acc[E_NUM];
#pragma unroll
    for (int e = 0; e < E_NUM; e++) acc[e] = 0.f;
    const float* xr = x + (size_t)row * K_DIM;
    for (int d = lane; d < K_DIM; d += 32) {
        const float xv = xr[d];
        const float* g = gw + (size_t)d * E_NUM;
#pragma unroll
        for (int e = 0; e < E_NUM; e++) acc[e] += xv * g[e];
    }
#pragma unroll
    for (int off = 16; off > 0; off >>= 1) {
#pragma unroll
        for (int e = 0; e < E_NUM; e++) acc[e] += __shfl_xor(acc[e], off, 32);
    }
    if (lane == 0) {
        float mx = -1e30f;
#pragma unroll
        for (int e = 0; e < E_NUM; e++) { acc[e] += gb[e]; mx = fmaxf(mx, acc[e]); }
        float s = 0.f;
#pragma unroll
        for (int e = 0; e < E_NUM; e++) { acc[e] = __expf(acc[e] - mx); s += acc[e]; }
        const float inv = 1.f / s;
#pragma unroll
        for (int e = 0; e < E_NUM; e++) gate[(size_t)row * E_NUM + e] = acc[e] * inv;
    }
}

// ---- fused MoE GEMM: out = (g .* x)_{M x 8192} @ W^T_{8192 x 1024} + g@b ----
__global__ __launch_bounds__(256) void moe_main(const float* __restrict__ x,
                                                const bf16_t* __restrict__ wbf,
                                                const float* __restrict__ gate,
                                                const float* __restrict__ eb,
                                                float* __restrict__ out) {
    __shared__ __align__(16) bf16_t As[2][BM][ASTR];     // gate-scaled x, row-major [m][k]
    __shared__ __align__(16) bf16_t Bs[2][BN][BSTR];     // W^T tile [n][k]
    __shared__ float gateS[BM][E_NUM];
    __shared__ float biasS[E_NUM][BN];

    const int tid = threadIdx.x;
    const int m0  = blockIdx.y * BM;
    const int n0  = blockIdx.x * BN;

    for (int i = tid; i < BM * E_NUM; i += 256)
        ((float*)gateS)[i] = gate[(size_t)m0 * E_NUM + i];
    for (int i = tid; i < E_NUM * BN; i += 256) {
        const int e = i / BN, c = i % BN;
        biasS[e][c] = eb[(size_t)e * N_DIM + n0 + c];
    }
    __syncthreads();

    v8f acc[2][4];
#pragma unroll
    for (int i = 0; i < 2; i++)
#pragma unroll
        for (int j = 0; j < 4; j++)
#pragma unroll
            for (int r = 0; r < 8; r++) acc[i][j][r] = 0.f;

    const int lane  = tid & 31;
    const int wave  = tid >> 5;
    const int wm    = wave >> 1;      // 0..3  (rows of 32)
    const int wn    = wave & 1;       // 0..1  (cols of 64)
    const int lhalf = lane >> 4;      // 0/1
    const int l16   = lane & 15;

    // A fill: gate-scale + f32->bf16, 2 threads per row, 32 elems each
    auto fillA = [&](int s, int kb) {
        const int e  = kb / KB_PER_E;                 // expert of this K'-slice
        const int kk = (kb % KB_PER_E) * BK;
        const int rA = tid >> 1;
        const int c0 = (tid & 1) * 32;
        const float g = gateS[rA][e];
        const float* src = x + (size_t)(m0 + rA) * K_DIM + kk + c0;
        union { bf16_t h[32]; uint4 q[4]; } u;
#pragma unroll
        for (int ii = 0; ii < 8; ii++) {
            float4 f = ((const float4*)src)[ii];
            u.h[4 * ii + 0] = (bf16_t)(f.x * g);
            u.h[4 * ii + 1] = (bf16_t)(f.y * g);
            u.h[4 * ii + 2] = (bf16_t)(f.z * g);
            u.h[4 * ii + 3] = (bf16_t)(f.w * g);
        }
#pragma unroll
        for (int ii = 0; ii < 4; ii++)
            *(uint4*)&As[s][rA][c0 + 8 * ii] = u.q[ii];
    };

    // B fill: [BN][BK] tile of pre-transposed W^T via TDM (one op per stage)
    auto fillB = [&](int s, int kb) {
        const int e  = kb / KB_PER_E;
        const int kk = (kb % KB_PER_E) * BK;
#if USE_TDM
        if (__builtin_amdgcn_readfirstlane(tid) == 0) {   // scalar guard: wave 0 only
            const unsigned lds_off = (unsigned)(size_t)&Bs[s][0][0];
            const unsigned long long ga =
                (unsigned long long)(const void*)(wbf + ((size_t)e * N_DIM + n0) * K_DIM + kk);
            v4u g0;
            g0[0] = 1u;                                   // count=1, user mode
            g0[1] = lds_off;                              // lds_addr
            g0[2] = (unsigned)ga;                         // global_addr[31:0]
            g0[3] = (unsigned)(ga >> 32) | (2u << 30);    // global_addr[56:32] | type=2
            v8i g1;
            g1[0] = (1 << 16)                             // data_size = 2B
                  | (1 << 20)                             // pad_enable
                  | (4 << 22)                             // pad_interval: 32 DWORDs
                  | (3 << 25);                            // pad_amount: 4 DWORDs
            g1[1] = (K_DIM & 0xFFFF) << 16;               // tensor_dim0[15:0]
            g1[2] = ((E_NUM * N_DIM) & 0xFFFF) << 16;     // tensor_dim1[15:0]
            g1[3] = BK << 16;                             // tile_dim0 = 64 (K, contiguous)
            g1[4] = BN;                                   // tile_dim1 = 128 rows
            g1[5] = K_DIM;                                // tensor_dim0_stride
            g1[6] = 0;
            g1[7] = 0;
            v4i g2 = {0, 0, 0, 0};
            v4i g3 = {0, 0, 0, 0};
#if __clang_major__ >= 23
            v8i g4 = {0, 0, 0, 0, 0, 0, 0, 0};
            __builtin_amdgcn_tensor_load_to_lds(g0, g1, g2, g3, g4, 0);
#else
            __builtin_amdgcn_tensor_load_to_lds(g0, g1, g2, g3, 0);
#endif
        }
#else
        const int rB = tid >> 1;            // n row 0..127
        const int c0 = (tid & 1) * 32;
        const bf16_t* src = wbf + ((size_t)e * N_DIM + n0 + rB) * K_DIM + kk + c0;
#pragma unroll
        for (int ii = 0; ii < 4; ii++)
            *(uint4*)&Bs[s][rB][c0 + 8 * ii] = ((const uint4*)src)[ii];
#endif
    };

    auto stage_sync = [&]() {
#if USE_TDM
        __builtin_amdgcn_s_wait_tensorcnt(0);   // free for non-issuing waves (cnt==0)
#endif
        __syncthreads();
    };

    fillB(0, 0);
    fillA(0, 0);
    stage_sync();

    for (int kb = 0; kb < NKB; kb++) {
        const int cur = kb & 1;
        if (kb + 1 < NKB) {
            fillB(cur ^ 1, kb + 1);
            fillA(cur ^ 1, kb + 1);
        }

#pragma unroll
        for (int ks = 0; ks < 2; ks++) {        // two WMMA k-steps per 64-deep stage
            const int kbase = ks * 32 + lhalf * 8;
            v16bf afrag[2], bfrag[4];
#pragma unroll
            for (int i = 0; i < 2; i++) {
                const int r = wm * 32 + i * 16 + l16;
                union { v16bf v; uint4 q[2]; } u;
                u.q[0] = *(const uint4*)&As[cur][r][kbase];
                u.q[1] = *(const uint4*)&As[cur][r][kbase + 16];
                afrag[i] = u.v;
            }
#pragma unroll
            for (int j = 0; j < 4; j++) {
                const int c = wn * 64 + j * 16 + l16;
                union { v16bf v; uint4 q[2]; } u;
                u.q[0] = *(const uint4*)&Bs[cur][c][kbase];
                u.q[1] = *(const uint4*)&Bs[cur][c][kbase + 16];
                bfrag[j] = u.v;
            }
#pragma unroll
            for (int i = 0; i < 2; i++)
#pragma unroll
                for (int j = 0; j < 4; j++)
                    acc[i][j] = __builtin_amdgcn_wmma_f32_16x16x32_bf16(
                        false, afrag[i], false, bfrag[j], (short)0, acc[i][j],
                        false, false);
        }
        stage_sync();
    }

    // epilogue: += sum_e g[m,e]*b[e,p], store f32
#pragma unroll
    for (int i = 0; i < 2; i++) {
#pragma unroll
        for (int j = 0; j < 4; j++) {
            const int cl  = wn * 64 + j * 16 + l16;
            const int col = n0 + cl;
#pragma unroll
            for (int r = 0; r < 8; r++) {
                const int rl = wm * 32 + i * 16 + lhalf * 8 + r;
                float v = acc[i][j][r];
#pragma unroll
                for (int e = 0; e < E_NUM; e++) v += gateS[rl][e] * biasS[e][cl];
                out[(size_t)(m0 + rl) * N_DIM + col] = v;
            }
        }
    }
}

extern "C" void kernel_launch(void* const* d_in, const int* in_sizes, int n_in,
                              void* d_out, int out_size, void* d_ws, size_t ws_size,
                              hipStream_t stream) {
    const float* x  = (const float*)d_in[0];
    const float* gw = (const float*)d_in[1];
    const float* gb = (const float*)d_in[2];
    const float* ew = (const float*)d_in[3];
    const float* eb = (const float*)d_in[4];
    float* out = (float*)d_out;

    bf16_t* wbf  = (bf16_t*)d_ws;                                             // 16 MB
    float*  gate = (float*)((char*)d_ws + (size_t)E_NUM * K_DIM * N_DIM * 2); // 256 KB

    dim3 tgrid(N_DIM / 32, K_DIM / 32, E_NUM);
    moe_convert_w_t<<<tgrid, 256, 0, stream>>>(ew, wbf);
    moe_gate<<<M_TOT / 8, 256, 0, stream>>>(x, gw, gb, gate);
    dim3 grid(N_DIM / BN, M_TOT / BM);
    moe_main<<<grid, 256, 0, stream>>>(x, wbf, gate, eb, out);
}
